// GraphTransformerLayer_34514357191069
// MI455X (gfx1250) — matrix operand order repaired and checked by reading, not compiled, not run on hardware
//
#include <hip/hip_runtime.h>

// ---------------------------------------------------------------------------
// GraphTransformerLayer on MI455X (gfx1250): bf16 WMMA pipeline.
// 32-row tiles, dual accumulator chains per wave (hides WMMA->WMMA hazards).
// ---------------------------------------------------------------------------

typedef __attribute__((ext_vector_type(16))) __bf16 v16bf;
typedef __attribute__((ext_vector_type(8)))  __bf16 v8bf;
typedef __attribute__((ext_vector_type(8)))  float  v8f;
typedef __attribute__((ext_vector_type(4)))  float  v4f;

#define DHD 128
#define AST 136   // LDS bf16 tile row stride (272B rows: bank-conflict pad)
#define FST 132   // LDS f32 tile row stride
#define FRAG_ELEMS 512          // 32 lanes * 16 bf16 per B fragment
#define FRAGS_PER_MAT 32        // 4 ktiles * 8 ntiles

__device__ __forceinline__ v8f wmma_bf16(v16bf a, v16bf b, v8f c) {
  return __builtin_amdgcn_wmma_f32_16x16x32_bf16(false, a, false, b, (short)0, c,
                                                 false, false);
}

// A fragment from a 32-row LDS tile; rowBase selects the 16-row half.
// Contiguous k order (as in B packing): lane l holds row rowBase+(l&15), K runs [klo,klo+16),
// klo = (l>>4)*16.
__device__ __forceinline__ v16bf load_a_frag(const __bf16* As, int rowBase,
                                             int lane, int kt) {
  const __bf16* p = As + (rowBase + (lane & 15)) * AST + kt * 32 +
                    ((lane >> 4) * 16);
  return *(const v16bf*)(p);
}

__device__ __forceinline__ v16bf load_b_frag(const __bf16* wfrag, int m, int kt,
                                             int nt, int lane) {
  return *(const v16bf*)(wfrag + ((size_t)((m * 4 + kt) * 8 + nt)) * FRAG_ELEMS +
                         lane * 16);
}

// dual-chain 128-deep GEMM for one 16-row half x two 16-col ntiles
__device__ __forceinline__ void gemm_dual(const __bf16* As, int rowBase, int lane,
                                          const __bf16* wfrag, int m, int nt0,
                                          v8f& acc0, v8f& acc1) {
#pragma unroll
  for (int kt = 0; kt < 4; kt++) {
    v16bf a = load_a_frag(As, rowBase, lane, kt);
    acc0 = wmma_bf16(a, load_b_frag(wfrag, m, kt, nt0, lane), acc0);
    acc1 = wmma_bf16(a, load_b_frag(wfrag, m, kt, nt0 + 1, lane), acc1);
  }
}

// ---------------------------------------------------------------------------
// Kernel 0: 5 DxD f32 weights -> bf16 WMMA B-fragment order.
// B layout (32x16): lane l -> col n = nt*16+(l&15), K = kt*32+(l>>4)*16+j.
// ---------------------------------------------------------------------------
__global__ __launch_bounds__(256) void swizzle_weights(
    const float* __restrict__ Wq, const float* __restrict__ Wk,
    const float* __restrict__ Wv, const float* __restrict__ We1,
    const float* __restrict__ Wo, __bf16* __restrict__ out) {
  int idx = blockIdx.x * blockDim.x + threadIdx.x;
  if (idx >= 5 * FRAGS_PER_MAT * FRAG_ELEMS) return;
  int j = idx & 15;
  int lane = (idx >> 4) & 31;
  int frag = (idx >> 9) & 31;  // kt*8 + nt
  int m = idx >> 14;
  int kt = frag >> 3, nt = frag & 7;
  int k = kt * 32 + (lane >> 4) * 16 + j;
  int n = nt * 16 + (lane & 15);
  const float* W = (m == 0) ? Wq : (m == 1) ? Wk : (m == 2) ? Wv
                   : (m == 3) ? We1 : Wo;
  out[idx] = (__bf16)W[k * DHD + n];
}

// ---------------------------------------------------------------------------
// Kernel 1: per-32-node tile: P = relu(c@Wp1+bp1)@Wp2+bp2 (in-tile), then
// Q = h@Wq, KP = h@Wk + P, VP = h@Wv + P, stored bf16.
// ---------------------------------------------------------------------------
__global__ __launch_bounds__(256) void node_proj(
    const float* __restrict__ h, const float* __restrict__ c,
    const float* __restrict__ Wp1, const float* __restrict__ bp1,
    const float* __restrict__ Wp2, const float* __restrict__ bp2,
    const __bf16* __restrict__ wfrag, __bf16* __restrict__ Qb,
    __bf16* __restrict__ KPb, __bf16* __restrict__ VPb, int N) {
  __shared__ __bf16 As[32 * AST];
  __shared__ __bf16 Cs[32 * AST];
  __shared__ float Ps[32 * FST];
  __shared__ float pt[32][4];

  int t = threadIdx.x;
  int lane = t & 31;
  int wave = t >> 5;
  int n0 = blockIdx.x * 32;
  int ti = t >> 3;            // 0..31 : tile row
  int td = (t & 7) * 16;      // 0..112: col chunk base (16 elems)

  // 32x128 h tile -> bf16 LDS (4x b128 load, 2x ds b128 store per thread)
  {
    int row = n0 + ti; if (row > N - 1) row = N - 1;
    const v4f* hp = (const v4f*)(h + (size_t)row * DHD + td);
    v4f a0 = hp[0], a1 = hp[1], a2 = hp[2], a3 = hp[3];
    v8bf r0, r1;
#pragma unroll
    for (int j = 0; j < 4; j++) {
      r0[j] = (__bf16)a0[j]; r0[4 + j] = (__bf16)a1[j];
      r1[j] = (__bf16)a2[j]; r1[4 + j] = (__bf16)a3[j];
    }
    *(v8bf*)(&As[ti * AST + td]) = r0;
    *(v8bf*)(&As[ti * AST + td + 8]) = r1;
  }
  // tiny MLP stage 1
  if (t < 32) {
    int row = n0 + t; if (row > N - 1) row = N - 1;
    float c0 = c[row * 3 + 0], c1 = c[row * 3 + 1], c2 = c[row * 3 + 2];
#pragma unroll
    for (int q = 0; q < 3; q++) {
      float v = c0 * Wp1[0 * 3 + q] + c1 * Wp1[1 * 3 + q] + c2 * Wp1[2 * 3 + q] +
                bp1[q];
      pt[t][q] = v > 0.f ? v : 0.f;
    }
  }
  __syncthreads();
  // stage 2: P tile (f32) in LDS
  {
    float t0 = pt[ti][0], t1 = pt[ti][1], t2 = pt[ti][2];
#pragma unroll
    for (int j = 0; j < 16; j++) {
      int d = td + j;
      Ps[ti * FST + d] = t0 * Wp2[0 * DHD + d] + t1 * Wp2[1 * DHD + d] +
                         t2 * Wp2[2 * DHD + d] + bp2[d];
    }
  }
  __syncthreads();

  int rowBase = (wave >> 2) * 16;          // 0 or 16
  int nt0 = (wave & 3) * 2;                // ntile pair
  int colLo = lane & 15;
  int rBase = (lane >> 4) * 8;
  int col0 = nt0 * 16 + colLo;
  bool rowOk = (n0 + ti) < N;
  size_t orow = (size_t)(n0 + ti) * DHD + td;

#pragma unroll
  for (int m = 0; m < 3; m++) {
    v8f acc0 = {}, acc1 = {};
    gemm_dual(As, rowBase, lane, wfrag, m, nt0, acc0, acc1);
#pragma unroll
    for (int r = 0; r < 8; r++) {
      int rr = rowBase + rBase + r;
      float v0 = acc0[r], v1 = acc1[r];
      if (m > 0) {
        v0 += Ps[rr * FST + col0];
        v1 += Ps[rr * FST + col0 + 16];
      }
      Cs[rr * AST + col0] = (__bf16)v0;
      Cs[rr * AST + col0 + 16] = (__bf16)v1;
    }
    __syncthreads();
    if (rowOk) {
      v8bf c0 = *(const v8bf*)(&Cs[ti * AST + td]);
      v8bf c1 = *(const v8bf*)(&Cs[ti * AST + td + 8]);
      __bf16* dstp = (m == 0) ? Qb : (m == 1) ? KPb : VPb;
      *(v8bf*)(dstp + orow) = c0;
      *(v8bf*)(dstp + orow + 8) = c1;
    }
    __syncthreads();
  }
}

// ---------------------------------------------------------------------------
// Kernel 2: per-32-edge tile: dif = KP[src]-Q[dst] -> bf16 LDS -> WMMA with
// We1 -> bias+relu -> dot We2 (shfl butterfly + LDS atomic) -> score ->
// scatter score*VP[src] into wV[dst] and score into z[dst] (f32 atomics).
// ---------------------------------------------------------------------------
__global__ __launch_bounds__(256) void edge_kernel(
    const int* __restrict__ src, const int* __restrict__ dst,
    const __bf16* __restrict__ Qb, const __bf16* __restrict__ KPb,
    const __bf16* __restrict__ VPb, const __bf16* __restrict__ wfrag,
    const float* __restrict__ be1, const float* __restrict__ We2,
    const float* __restrict__ be2, float* __restrict__ wV,
    float* __restrict__ z, int E) {
  __shared__ __bf16 As[32 * AST];
  __shared__ int sSrc[32], sDst[32];
  __shared__ float sRaw[32];
  __shared__ float sScore[32];

  int t = threadIdx.x;
  int lane = t & 31;
  int wave = t >> 5;
  int e0 = blockIdx.x * 32;
  int ti = t >> 3;            // 0..31 : edge within tile
  int td = (t & 7) * 16;      // col chunk (16 elems)

  if (t < 32) {
    int e = e0 + t; if (e > E - 1) e = E - 1;
    sSrc[t] = src[e];
    sDst[t] = dst[e];
    sRaw[t] = 0.f;
  }
  __syncthreads();
  // gather dif rows: 4x b128 load, f32 subtract, 2x ds b128 store
  {
    int s = sSrc[ti], d = sDst[ti];
    const v8bf* kp = (const v8bf*)(KPb + (size_t)s * DHD + td);
    const v8bf* qp = (const v8bf*)(Qb + (size_t)d * DHD + td);
    v8bf k0 = kp[0], k1 = kp[1], q0 = qp[0], q1 = qp[1];
    v8bf r0, r1;
#pragma unroll
    for (int j = 0; j < 8; j++) {
      r0[j] = (__bf16)((float)k0[j] - (float)q0[j]);
      r1[j] = (__bf16)((float)k1[j] - (float)q1[j]);
    }
    *(v8bf*)(&As[ti * AST + td]) = r0;
    *(v8bf*)(&As[ti * AST + td + 8]) = r1;
  }
  __syncthreads();

  int rowBase = (wave >> 2) * 16;
  int nt0 = (wave & 3) * 2;
  int colLo = lane & 15;
  int rBase = (lane >> 4) * 8;
  int col0 = nt0 * 16 + colLo;

  v8f acc0 = {}, acc1 = {};
  gemm_dual(As, rowBase, lane, wfrag, 3, nt0, acc0, acc1);

  float w20 = We2[col0], w21 = We2[col0 + 16];
  float b0 = be1[col0], b1 = be1[col0 + 16];
#pragma unroll
  for (int r = 0; r < 8; r++) {
    float h0 = acc0[r] + b0; h0 = h0 > 0.f ? h0 : 0.f;
    float h1 = acc1[r] + b1; h1 = h1 > 0.f ? h1 : 0.f;
    float p = h0 * w20 + h1 * w21;
#pragma unroll
    for (int mask = 1; mask < 16; mask <<= 1) p += __shfl_xor(p, mask, 32);
    if ((lane & 15) == 0) atomicAdd(&sRaw[rowBase + rBase + r], p);
  }
  __syncthreads();

  if (t < 32) {
    float raw = (sRaw[t] + be2[0]) * 0.08838834764831845f;  // 1/sqrt(128)
    raw = fminf(fmaxf(raw, -5.f), 5.f);
    float sc = __expf(raw);
    sScore[t] = sc;
    if (e0 + t < E) atomicAdd(&z[sDst[t]], sc);
  }
  __syncthreads();

  if (e0 + ti < E) {
    int s = sSrc[ti], d = sDst[ti];
    float sc = sScore[ti];
    const v8bf* vp = (const v8bf*)(VPb + (size_t)s * DHD + td);
    v8bf v0 = vp[0], v1 = vp[1];
    float* o = wV + (size_t)d * DHD + td;
#pragma unroll
    for (int j = 0; j < 8; j++) {
      atomicAdd(&o[j], (float)v0[j] * sc);
      atomicAdd(&o[j + 8], (float)v1[j] * sc);
    }
  }
}

// ---------------------------------------------------------------------------
// Kernel 3: out = (wV / z) @ Wo + bo
// ---------------------------------------------------------------------------
__global__ __launch_bounds__(256) void out_proj(
    const float* __restrict__ wV, const float* __restrict__ z,
    const __bf16* __restrict__ wfrag, const float* __restrict__ bo,
    float* __restrict__ out, int N) {
  __shared__ __bf16 As[32 * AST];
  __shared__ float Cs[32 * FST];
  int t = threadIdx.x;
  int lane = t & 31;
  int wave = t >> 5;
  int n0 = blockIdx.x * 32;
  int ti = t >> 3;
  int td = (t & 7) * 16;

  {
    int row = n0 + ti; if (row > N - 1) row = N - 1;
    float inv = 1.0f / z[row];
    const v4f* wp = (const v4f*)(wV + (size_t)row * DHD + td);
    v4f a0 = wp[0], a1 = wp[1], a2 = wp[2], a3 = wp[3];
    v8bf r0, r1;
#pragma unroll
    for (int j = 0; j < 4; j++) {
      r0[j] = (__bf16)(a0[j] * inv); r0[4 + j] = (__bf16)(a1[j] * inv);
      r1[j] = (__bf16)(a2[j] * inv); r1[4 + j] = (__bf16)(a3[j] * inv);
    }
    *(v8bf*)(&As[ti * AST + td]) = r0;
    *(v8bf*)(&As[ti * AST + td + 8]) = r1;
  }
  __syncthreads();

  int rowBase = (wave >> 2) * 16;
  int nt0 = (wave & 3) * 2;
  int colLo = lane & 15;
  int rBase = (lane >> 4) * 8;
  int col0 = nt0 * 16 + colLo;

  v8f acc0 = {}, acc1 = {};
  gemm_dual(As, rowBase, lane, wfrag, 4, nt0, acc0, acc1);

  float b0 = bo[col0], b1 = bo[col0 + 16];
#pragma unroll
  for (int r = 0; r < 8; r++) {
    int rr = rowBase + rBase + r;
    Cs[rr * FST + col0] = acc0[r] + b0;
    Cs[rr * FST + col0 + 16] = acc1[r] + b1;
  }
  __syncthreads();

  if (n0 + ti < N) {
    const v4f* cp = (const v4f*)(&Cs[ti * FST + td]);
    v4f c0 = cp[0], c1 = cp[1], c2 = cp[2], c3 = cp[3];
    float* op = out + (size_t)(n0 + ti) * DHD + td;
    *(v4f*)(op) = c0;
    *(v4f*)(op + 4) = c1;
    *(v4f*)(op + 8) = c2;
    *(v4f*)(op + 12) = c3;
  }
}

// ---------------------------------------------------------------------------
static inline size_t alignup256(size_t x) { return (x + 255) & ~(size_t)255; }

extern "C" void kernel_launch(void* const* d_in, const int* in_sizes, int n_in,
                              void* d_out, int out_size, void* d_ws,
                              size_t ws_size, hipStream_t stream) {
  (void)n_in; (void)out_size; (void)ws_size;
  const float* h   = (const float*)d_in[0];
  const float* c   = (const float*)d_in[1];
  const int*   src = (const int*)d_in[2];
  const int*   dst = (const int*)d_in[3];
  const float* Wq  = (const float*)d_in[4];
  const float* Wk  = (const float*)d_in[5];
  const float* Wv  = (const float*)d_in[6];
  const float* Wp1 = (const float*)d_in[7];
  const float* bp1 = (const float*)d_in[8];
  const float* Wp2 = (const float*)d_in[9];
  const float* bp2 = (const float*)d_in[10];
  const float* We1 = (const float*)d_in[11];
  const float* be1 = (const float*)d_in[12];
  const float* We2 = (const float*)d_in[13];
  const float* be2 = (const float*)d_in[14];
  const float* Wo  = (const float*)d_in[15];
  const float* bo  = (const float*)d_in[16];
  float* out = (float*)d_out;

  int N = in_sizes[0] / DHD;
  int E = in_sizes[2];
  size_t ND = (size_t)N * DHD;

  // workspace carve
  char* w = (char*)d_ws;
  __bf16* Qb  = (__bf16*)w; w += alignup256(ND * sizeof(__bf16));
  __bf16* KPb = (__bf16*)w; w += alignup256(ND * sizeof(__bf16));
  __bf16* VPb = (__bf16*)w; w += alignup256(ND * sizeof(__bf16));
  float*  wV  = (float*)w;  w += alignup256(ND * sizeof(float));
  float*  z   = (float*)w;  w += alignup256((size_t)N * sizeof(float));
  __bf16* wfrag = (__bf16*)w;  // 5 * 32 * 512 bf16

  hipMemsetAsync(wV, 0, ND * sizeof(float), stream);
  hipMemsetAsync(z, 0, (size_t)N * sizeof(float), stream);

  {
    int total = 5 * FRAGS_PER_MAT * FRAG_ELEMS;
    swizzle_weights<<<(total + 255) / 256, 256, 0, stream>>>(Wq, Wk, Wv, We1, Wo,
                                                             wfrag);
  }
  node_proj<<<(N + 31) / 32, 256, 0, stream>>>(h, c, Wp1, bp1, Wp2, bp2, wfrag,
                                               Qb, KPb, VPb, N);
  edge_kernel<<<(E + 31) / 32, 256, 0, stream>>>(src, dst, Qb, KPb, VPb, wfrag,
                                                 be1, We2, be2, wV, z, E);
  out_proj<<<(N + 31) / 32, 256, 0, stream>>>(wV, z, wfrag, bo, out, N);
}
